// RingCrossAttention_78228534330090
// MI455X (gfx1250) — matrix operand order repaired and checked by reading, not compiled, not run on hardware
//
#include <hip/hip_runtime.h>

#define HID 1024
#define NH  16
#define HD  64
#define BSZ 2
#define SQn 2048
#define SKV 4096
#define SCALE 0.125f   // 1/sqrt(64)

typedef __attribute__((ext_vector_type(16))) __bf16 v16bf;
typedef __attribute__((ext_vector_type(8)))  float  v8f;

union FragB16 {
    v16bf          v;
    unsigned short u[16];
    uint4          q[2];
};

static __device__ inline unsigned short f2bf(float f) {
    __bf16 h = (__bf16)f;                       // hardware cvt, RNE
    return __builtin_bit_cast(unsigned short, h);
}

static __device__ inline v8f zero_v8f() {
    v8f z;
#pragma unroll
    for (int i = 0; i < 8; ++i) z[i] = 0.0f;
    return z;
}

static __device__ inline v8f wmma_bf16(const FragB16& a, const FragB16& b, v8f c) {
    return __builtin_amdgcn_wmma_f32_16x16x32_bf16(
        false, a.v, false, b.v, (short)0, c, false, false);
}

// One async 16B copy global->LDS per lane (VGLOBAL GLOBAL_LOAD_ASYNC_TO_LDS_B128).
static __device__ inline void async_b128(unsigned short* lds, const unsigned short* g) {
    unsigned           loff = (unsigned)(unsigned long long)(uintptr_t)lds;  // LDS byte offset
    unsigned long long ga   = (unsigned long long)(uintptr_t)g;
    asm volatile("global_load_async_to_lds_b128 %0, %1, off"
                 :: "v"(loff), "v"(ga) : "memory");
}
#define WAIT_ASYNC(n) asm volatile("s_wait_asynccnt %0" :: "i"(n) : "memory")
#define WAIT_DS0()    asm volatile("s_wait_dscnt 0" ::: "memory")

// ---------------------------------------------------------------------------
// fp32 -> bf16 bulk conversion (bandwidth-bound, done once so GEMM hot loops
// carry zero conversion VALU).  n8 = element count / 8.
// ---------------------------------------------------------------------------
__global__ __launch_bounds__(256) void cvt_kernel(
    const float* __restrict__ s, unsigned short* __restrict__ d, int n8)
{
    int i      = blockIdx.x * 256 + threadIdx.x;
    int stride = gridDim.x * 256;
    for (; i < n8; i += stride) {
        const float4* sp = reinterpret_cast<const float4*>(s) + (size_t)i * 2;
        float4 a = sp[0], b = sp[1];
        uint4 o;
        o.x = (unsigned)f2bf(a.x) | ((unsigned)f2bf(a.y) << 16);
        o.y = (unsigned)f2bf(a.z) | ((unsigned)f2bf(a.w) << 16);
        o.z = (unsigned)f2bf(b.x) | ((unsigned)f2bf(b.y) << 16);
        o.w = (unsigned)f2bf(b.z) | ((unsigned)f2bf(b.w) << 16);
        reinterpret_cast<uint4*>(d)[i] = o;
    }
}

// ---------------------------------------------------------------------------
// Projection GEMM (all-bf16 operands): Y[m,n] = sum_k X[m,k]*W[n,k] + bias[n]
// Block = 128 threads (4 waves); block tile 64(M) x 128(N).
// Each wave: two 16-col N strips (n0, n0+64) x four 16-row M subtiles
//  -> 8 independent accumulators, A fragment reused for 2 WMMAs.
// MODE 0: Y bf16 [b][h][sq][d] * SCALE   (Q)
// MODE 1: Y bf16 [b][h][skv][d]          (K)
// MODE 2: Y bf16 [b][h][d][skv]          (V, transposed)
// MODE 3: Y f32  [m][n] (d_out)          (O)
// ---------------------------------------------------------------------------
template <int MODE>
__global__ __launch_bounds__(128) void proj_kernel(
    const unsigned short* __restrict__ X, const unsigned short* __restrict__ W,
    const float* __restrict__ bias,
    unsigned short* __restrict__ Yh, float* __restrict__ Yf)
{
    const int lane = threadIdx.x & 31;
    const int wave = threadIdx.x >> 5;
    const int l16  = lane & 15;
    const int lh   = lane >> 4;

    const int m_base = blockIdx.x * 64;
    const int nA     = blockIdx.y * 128 + wave * 16 + l16;
    const int nB     = nA + 64;

    const unsigned short* wrowA = W + (size_t)nA * HID;
    const unsigned short* wrowB = W + (size_t)nB * HID;

    v8f acc[2][4];
#pragma unroll
    for (int s = 0; s < 2; ++s)
#pragma unroll
        for (int i = 0; i < 4; ++i) acc[s][i] = zero_v8f();

    for (int k0 = 0; k0 < HID; k0 += 32) {
        FragB16 bfA, bfB;
        bfA.q[0] = *reinterpret_cast<const uint4*>(wrowA + k0 + 16 * lh);
        bfA.q[1] = *reinterpret_cast<const uint4*>(wrowA + k0 + 16 * lh + 8);
        bfB.q[0] = *reinterpret_cast<const uint4*>(wrowB + k0 + 16 * lh);
        bfB.q[1] = *reinterpret_cast<const uint4*>(wrowB + k0 + 16 * lh + 8);
        if (k0 + 32 < HID)
            __builtin_prefetch(wrowA + k0 + 32 + 16 * lh, 0, 3);

#pragma unroll
        for (int i = 0; i < 4; ++i) {
            const unsigned short* xr = X + (size_t)(m_base + 16 * i + l16) * HID + k0;
            FragB16 af;
            af.q[0] = *reinterpret_cast<const uint4*>(xr + 16 * lh);
            af.q[1] = *reinterpret_cast<const uint4*>(xr + 16 * lh + 8);
            acc[0][i] = wmma_bf16(af, bfA, acc[0][i]);
            acc[1][i] = wmma_bf16(af, bfB, acc[1][i]);
        }
    }

#pragma unroll
    for (int s = 0; s < 2; ++s) {
        const int   n  = s ? nB : nA;
        const float bn = bias[n];
        const int   h  = n >> 6;
        const int   d  = n & 63;
#pragma unroll
        for (int i = 0; i < 4; ++i) {
#pragma unroll
            for (int r = 0; r < 8; ++r) {
                const int m = m_base + 16 * i + 8 * lh + r;
                float y = acc[s][i][r] + bn;
                if (MODE == 0) {
                    y *= SCALE;
                    const int b = m >> 11, sq = m & 2047;
                    Yh[((((size_t)b * NH + h) * SQn + sq) << 6) + d] = f2bf(y);
                } else if (MODE == 1) {
                    const int b = m >> 12, skv = m & 4095;
                    Yh[((((size_t)b * NH + h) * SKV + skv) << 6) + d] = f2bf(y);
                } else if (MODE == 2) {
                    const int b = m >> 12, skv = m & 4095;
                    Yh[(((size_t)(b * NH + h) * HD + d) * SKV) + skv] = f2bf(y);
                } else {
                    Yf[(size_t)m * HID + n] = y;
                }
            }
        }
    }
}

// ---------------------------------------------------------------------------
// Flash attention.  Block = 4 waves, all on the same (b,h); each wave owns a
// 32-query pair of tiles.  K/V 32-key chunks are double-buffered in LDS via
// global_load_async_to_lds_b128 (s_wait_asynccnt pipelined).  16 WMMA per
// chunk per wave; P goes D-layout -> A-layout through LDS.
// ---------------------------------------------------------------------------
__global__ __launch_bounds__(128) void attn_kernel(
    const unsigned short* __restrict__ qh,   // [b][h][sq][d]  bf16 (pre-scaled)
    const unsigned short* __restrict__ kh,   // [b][h][skv][d] bf16
    const unsigned short* __restrict__ vt,   // [b][h][d][skv] bf16
    unsigned short* __restrict__ ctx)        // [b][sq][h*64+d] bf16
{
    __shared__ unsigned short sK[2][32 * 64];        // 2 x 4KB: [key_local][d]
    __shared__ unsigned short sV[2][64 * 32];        // 2 x 4KB: [d][key_local]
    __shared__ unsigned short sP[4][2][16 * 32];     // 8KB: [wave][qt][q][key]

    const int tid  = threadIdx.x;
    const int lane = tid & 31;
    const int wave = tid >> 5;
    const int l16  = lane & 15;
    const int lh   = lane >> 4;

    const int tp = blockIdx.x * 4 + wave;   // 32-query pair id, [0, 2048)
    const int bh = tp >> 6;                 // same for all 4 waves of a block
    const int q0 = (tp & 63) << 5;
    const int b  = bh >> 4;
    const int h  = bh & 15;

    const unsigned short* kbase = kh + (size_t)bh * SKV * HD;
    const unsigned short* vbase = vt + (size_t)bh * HD * SKV;

    // Persistent Q A-fragments: [qt][d-chunk]
    FragB16 aq[2][2];
#pragma unroll
    for (int qt = 0; qt < 2; ++qt) {
        const unsigned short* qrow =
            qh + ((size_t)bh * SQn + q0 + qt * 16 + l16) * HD;
        aq[qt][0].q[0] = *reinterpret_cast<const uint4*>(qrow + 16 * lh);
        aq[qt][0].q[1] = *reinterpret_cast<const uint4*>(qrow + 16 * lh + 8);
        aq[qt][1].q[0] = *reinterpret_cast<const uint4*>(qrow + 32 + 16 * lh);
        aq[qt][1].q[1] = *reinterpret_cast<const uint4*>(qrow + 32 + 16 * lh + 8);
    }

    v8f cacc[2][4];
#pragma unroll
    for (int qt = 0; qt < 2; ++qt)
#pragma unroll
        for (int j = 0; j < 4; ++j) cacc[qt][j] = zero_v8f();
    float rmax[2][8], rsum[2][8];
#pragma unroll
    for (int qt = 0; qt < 2; ++qt)
#pragma unroll
        for (int r = 0; r < 8; ++r) { rmax[qt][r] = -1e30f; rsum[qt][r] = 0.0f; }

    // cooperative async staging of one 32-key chunk (4 x b128 per thread)
    auto issue_chunk = [&](int key0, int buf) {
        const unsigned short* gk = kbase + (size_t)key0 * HD;  // 4KB contiguous
        const int e0 = tid * 8, e1 = (tid + 128) * 8;
        async_b128(&sK[buf][e0], gk + e0);
        async_b128(&sK[buf][e1], gk + e1);
        const int i0 = tid * 2, i1 = i0 + 1;
        const int r0 = i0 >> 2, g0 = i0 & 3;
        const int r1 = i1 >> 2, g1 = i1 & 3;
        async_b128(&sV[buf][r0 * 32 + g0 * 8], vbase + (size_t)r0 * SKV + key0 + g0 * 8);
        async_b128(&sV[buf][r1 * 32 + g1 * 8], vbase + (size_t)r1 * SKV + key0 + g1 * 8);
    };

    issue_chunk(0, 0);
    const int NCH = SKV / 32;
    for (int it = 0; it < NCH; ++it) {
        const int buf = it & 1;
        if (it + 1 < NCH) {
            issue_chunk((it + 1) * 32, buf ^ 1);
            WAIT_ASYNC(4);                 // oldest 4 (this chunk) complete
        } else {
            WAIT_ASYNC(0);
        }
        __syncthreads();                   // all waves' stages visible

        // ---------------- scores: 8 WMMA, interleaved accumulators ----------
        v8f s[2][2];
#pragma unroll
        for (int qt = 0; qt < 2; ++qt)
#pragma unroll
            for (int kt = 0; kt < 2; ++kt) s[qt][kt] = zero_v8f();
#pragma unroll
        for (int kt = 0; kt < 2; ++kt) {
            const unsigned short* kr = &sK[buf][(kt * 16 + l16) * HD + 16 * lh];
            FragB16 kb0, kb1;
            kb0.q[0] = *reinterpret_cast<const uint4*>(kr);
            kb0.q[1] = *reinterpret_cast<const uint4*>(kr + 8);
            kb1.q[0] = *reinterpret_cast<const uint4*>(kr + 32);
            kb1.q[1] = *reinterpret_cast<const uint4*>(kr + 40);
            s[0][kt] = wmma_bf16(aq[0][0], kb0, s[0][kt]);
            s[1][kt] = wmma_bf16(aq[1][0], kb0, s[1][kt]);
            s[0][kt] = wmma_bf16(aq[0][1], kb1, s[0][kt]);
            s[1][kt] = wmma_bf16(aq[1][1], kb1, s[1][kt]);
        }

        // ---------------- online softmax per query tile ----------------------
#pragma unroll
        for (int qt = 0; qt < 2; ++qt) {
#pragma unroll
            for (int r = 0; r < 8; ++r) {
                float mx = fmaxf(s[qt][0][r], s[qt][1][r]);
                mx = fmaxf(mx, __shfl_xor(mx, 1, 16));
                mx = fmaxf(mx, __shfl_xor(mx, 2, 16));
                mx = fmaxf(mx, __shfl_xor(mx, 4, 16));
                mx = fmaxf(mx, __shfl_xor(mx, 8, 16));
                const float nm   = fmaxf(rmax[qt][r], mx);
                const float corr = __expf(rmax[qt][r] - nm);
                rmax[qt][r] = nm;
                const float p0 = __expf(s[qt][0][r] - nm);
                const float p1 = __expf(s[qt][1][r] - nm);
                float ps = p0 + p1;
                ps += __shfl_xor(ps, 1, 16);
                ps += __shfl_xor(ps, 2, 16);
                ps += __shfl_xor(ps, 4, 16);
                ps += __shfl_xor(ps, 8, 16);
                rsum[qt][r] = rsum[qt][r] * corr + ps;
#pragma unroll
                for (int j = 0; j < 4; ++j) cacc[qt][j][r] *= corr;
                const int row = r + 8 * lh;
                sP[wave][qt][row * 32 + l16]      = f2bf(p0);
                sP[wave][qt][row * 32 + 16 + l16] = f2bf(p1);
            }
        }
        WAIT_DS0();   // same-wave LDS ordering for the P round-trip

        // ---------------- P.V: 8 WMMA, shared V fragments --------------------
        FragB16 ap[2];
#pragma unroll
        for (int qt = 0; qt < 2; ++qt) {
            ap[qt].q[0] = *reinterpret_cast<const uint4*>(&sP[wave][qt][l16 * 32 + 16 * lh]);
            ap[qt].q[1] = *reinterpret_cast<const uint4*>(&sP[wave][qt][l16 * 32 + 16 * lh + 8]);
        }
#pragma unroll
        for (int j = 0; j < 4; ++j) {
            const unsigned short* vr = &sV[buf][(16 * j + l16) * 32 + 16 * lh];
            FragB16 vb;
            vb.q[0] = *reinterpret_cast<const uint4*>(vr);
            vb.q[1] = *reinterpret_cast<const uint4*>(vr + 8);
            cacc[0][j] = wmma_bf16(ap[0], vb, cacc[0][j]);
            cacc[1][j] = wmma_bf16(ap[1], vb, cacc[1][j]);
        }
        __syncthreads();   // everyone done with buf before it is re-staged
    }

    // ---------------- finalize ------------------------------------------------
#pragma unroll
    for (int qt = 0; qt < 2; ++qt) {
#pragma unroll
        for (int r = 0; r < 8; ++r) {
            const float  inv = 1.0f / rsum[qt][r];
            const size_t qg  = (size_t)b * SQn + q0 + qt * 16 + r + 8 * lh;
#pragma unroll
            for (int j = 0; j < 4; ++j)
                ctx[qg * HID + h * HD + 16 * j + l16] = f2bf(cacc[qt][j][r] * inv);
        }
    }
}

// ---------------------------------------------------------------------------
extern "C" void kernel_launch(void* const* d_in, const int* in_sizes, int n_in,
                              void* d_out, int out_size, void* d_ws, size_t ws_size,
                              hipStream_t stream) {
    (void)in_sizes; (void)n_in; (void)out_size; (void)ws_size;
    const float* Xq  = (const float*)d_in[0];
    const float* Xkv = (const float*)d_in[1];
    const float* Wq  = (const float*)d_in[2];
    const float* bq  = (const float*)d_in[3];
    const float* Wk  = (const float*)d_in[4];
    const float* bk  = (const float*)d_in[5];
    const float* Wv  = (const float*)d_in[6];
    const float* bv  = (const float*)d_in[7];
    const float* Wo  = (const float*)d_in[8];
    const float* bo  = (const float*)d_in[9];
    float* out = (float*)d_out;

    // workspace (bf16 elems): qh 4M | kh 8M | vt 8M | ctx 4M | xq 4M | xkv 8M
    //                         | wq 1M | wk 1M | wv 1M | wo 1M   = 40M = 80 MB
    unsigned short* qhp  = (unsigned short*)d_ws;
    unsigned short* khp  = qhp  + (size_t)BSZ * NH * SQn * HD;
    unsigned short* vtp  = khp  + (size_t)BSZ * NH * SKV * HD;
    unsigned short* ctp  = vtp  + (size_t)BSZ * NH * SKV * HD;
    unsigned short* xqh  = ctp  + (size_t)BSZ * SQn * HID;
    unsigned short* xkvh = xqh  + (size_t)BSZ * SQn * HID;
    unsigned short* wqh  = xkvh + (size_t)BSZ * SKV * HID;
    unsigned short* wkh  = wqh  + (size_t)HID * HID;
    unsigned short* wvh  = wkh  + (size_t)HID * HID;
    unsigned short* woh  = wvh  + (size_t)HID * HID;

    const dim3 blk(128), cblk(256);
    const int nXq = (BSZ * SQn * HID) / 8, nXkv = (BSZ * SKV * HID) / 8;
    const int nW  = (HID * HID) / 8;
    cvt_kernel<<<dim3(1024), cblk, 0, stream>>>(Xq,  xqh,  nXq);
    cvt_kernel<<<dim3(2048), cblk, 0, stream>>>(Xkv, xkvh, nXkv);
    cvt_kernel<<<dim3(512),  cblk, 0, stream>>>(Wq,  wqh,  nW);
    cvt_kernel<<<dim3(512),  cblk, 0, stream>>>(Wk,  wkh,  nW);
    cvt_kernel<<<dim3(512),  cblk, 0, stream>>>(Wv,  wvh,  nW);
    cvt_kernel<<<dim3(512),  cblk, 0, stream>>>(Wo,  woh,  nW);

    proj_kernel<0><<<dim3((BSZ * SQn) / 64, HID / 128), blk, 0, stream>>>(
        xqh, wqh, bq, qhp, nullptr);
    proj_kernel<1><<<dim3((BSZ * SKV) / 64, HID / 128), blk, 0, stream>>>(
        xkvh, wkh, bk, khp, nullptr);
    proj_kernel<2><<<dim3((BSZ * SKV) / 64, HID / 128), blk, 0, stream>>>(
        xkvh, wvh, bv, vtp, nullptr);

    attn_kernel<<<dim3((BSZ * NH * SQn) / 128), blk, 0, stream>>>(qhp, khp, vtp, ctp);

    proj_kernel<3><<<dim3((BSZ * SQn) / 64, HID / 128), blk, 0, stream>>>(
        ctp, woh, bo, nullptr, out);
}
